// GCNConv_12232066859482
// MI455X (gfx1250) — matrix operand order repaired and checked by reading, not compiled, not run on hardware
//
#include <hip/hip_runtime.h>
#include <hip/hip_bf16.h>

typedef __attribute__((ext_vector_type(16))) _Float16 f16x16;
typedef __attribute__((ext_vector_type(8)))  _Float16 f16x8;
typedef __attribute__((ext_vector_type(8)))  float    f32x8;

#define CEIL_DIV(a, b) (((a) + (b) - 1) / (b))

// ---------------------------------------------------------------------------
// Pack W (K x Oc, f32 row-major) into WMMA B-fragment order (f16):
// idx = ((ct*ksteps + ks)*32 + lane)*16 + e
//   col = ct*16 + (lane&15); k = ks*32 + 16*(e>>3) + 8*(lane>>4) + (e&7)
// ---------------------------------------------------------------------------
__global__ void pack_w_kernel(const float* __restrict__ W, _Float16* __restrict__ Wp,
                              int K, int Oc) {
    int ksteps = K >> 5;
    int total  = K * Oc;
    int idx = blockIdx.x * blockDim.x + threadIdx.x;
    if (idx >= total) return;
    int e    = idx & 15;
    int lane = (idx >> 4) & 31;
    int ks   = (idx >> 9) % ksteps;
    int ct   = idx / (ksteps << 9);
    int col  = (ct << 4) + (lane & 15);
    int k    = (ks << 5) + ((e >> 3) << 4) + ((lane >> 4) << 3) + (e & 7);
    Wp[idx]  = (_Float16)W[(size_t)k * Oc + col];
}

// ---------------------------------------------------------------------------
// f32 -> f16 activation transform: optional per-column BN affine + leaky relu
// ---------------------------------------------------------------------------
__global__ void xform_f16_kernel(const float* __restrict__ x,
                                 const float* __restrict__ scale,
                                 const float* __restrict__ shift,
                                 int apply_lrelu,
                                 _Float16* __restrict__ out, int total, int F) {
    int i = blockIdx.x * blockDim.x + threadIdx.x;
    if (i >= total) return;
    float v = x[i];
    if (scale) {
        int f = i % F;
        v = v * scale[f] + shift[f];
    }
    if (apply_lrelu) v = v > 0.0f ? v : 0.01f * v;
    out[i] = (_Float16)v;
}

// ---------------------------------------------------------------------------
// WMMA GEMM: C[N x Oc] = Ah[N x K](f16) * Wp(packed f16) (+ bias)
// One wave computes a 32 x 64 block (2 row-tiles x 4 col-tiles): per K-step,
// 4 A loads + 4 B loads feed 8 wmmas. N%16==0, K%32==0, Oc%64==0.
// ---------------------------------------------------------------------------
__global__ __launch_bounds__(256) void gemm_wmma_f16(
    const _Float16* __restrict__ Ah, const _Float16* __restrict__ Wp,
    const float* __restrict__ bias, float* __restrict__ C,
    int N, int K, int Oc) {
    int lane = threadIdx.x & 31;
    int wave = threadIdx.x >> 5;
    int grp  = blockIdx.x * (blockDim.x >> 5) + wave;   // tile-group id
    int MT  = N >> 4;                                    // 16-row tiles
    int MG  = (MT + 1) >> 1;                             // groups of 2 row-tiles
    int OCG = Oc >> 6;                                   // groups of 4 col-tiles
    if (grp >= MG * OCG) return;
    int mg = grp / OCG;
    int cg = grp - mg * OCG;
    int mt0 = mg << 1;
    int mt1 = mt0 + 1;
    bool has1 = mt1 < MT;
    int mt1c = has1 ? mt1 : mt0;                         // clamp (loads only)
    int ksteps = K >> 5;
    int khalf  = lane >> 4;                              // 0 or 1

    // A: row = mt*16 + (lane&15); elem 0..7 -> K = ks*32 + 8*khalf + e
    //                             elem 8..15 -> K = ks*32 + 16 + 8*khalf + e
    const _Float16* arow0 = Ah + (size_t)((mt0 << 4) + (lane & 15)) * K + (khalf << 3);
    const _Float16* arow1 = Ah + (size_t)((mt1c << 4) + (lane & 15)) * K + (khalf << 3);
    // B fragments for col-tiles ct = cg*4 + c
    const _Float16* wp = Wp + (((size_t)(cg << 2) * ksteps) << 9) + lane * 16;
    const size_t ctStride = (size_t)ksteps << 9;         // elems between col-tiles

    f32x8 acc00 = {}, acc01 = {}, acc02 = {}, acc03 = {};
    f32x8 acc10 = {}, acc11 = {}, acc12 = {}, acc13 = {};
    for (int ks = 0; ks < ksteps; ++ks) {
        f16x8 lo0 = *(const f16x8*)(arow0 + (ks << 5));
        f16x8 hi0 = *(const f16x8*)(arow0 + (ks << 5) + 16);
        f16x8 lo1 = *(const f16x8*)(arow1 + (ks << 5));
        f16x8 hi1 = *(const f16x8*)(arow1 + (ks << 5) + 16);
        f16x16 a0, a1;
#pragma unroll
        for (int e = 0; e < 8; ++e) {
            a0[e] = lo0[e]; a0[e + 8] = hi0[e];
            a1[e] = lo1[e]; a1[e + 8] = hi1[e];
        }
        const _Float16* wk = wp + ((size_t)ks << 9);
        f16x16 b0 = *(const f16x16*)(wk);
        f16x16 b1 = *(const f16x16*)(wk + ctStride);
        f16x16 b2 = *(const f16x16*)(wk + 2 * ctStride);
        f16x16 b3 = *(const f16x16*)(wk + 3 * ctStride);
        acc00 = __builtin_amdgcn_wmma_f32_16x16x32_f16(false, a0, false, b0, (short)0, acc00, false, false);
        acc10 = __builtin_amdgcn_wmma_f32_16x16x32_f16(false, a1, false, b0, (short)0, acc10, false, false);
        acc01 = __builtin_amdgcn_wmma_f32_16x16x32_f16(false, a0, false, b1, (short)0, acc01, false, false);
        acc11 = __builtin_amdgcn_wmma_f32_16x16x32_f16(false, a1, false, b1, (short)0, acc11, false, false);
        acc02 = __builtin_amdgcn_wmma_f32_16x16x32_f16(false, a0, false, b2, (short)0, acc02, false, false);
        acc12 = __builtin_amdgcn_wmma_f32_16x16x32_f16(false, a1, false, b2, (short)0, acc12, false, false);
        acc03 = __builtin_amdgcn_wmma_f32_16x16x32_f16(false, a0, false, b3, (short)0, acc03, false, false);
        acc13 = __builtin_amdgcn_wmma_f32_16x16x32_f16(false, a1, false, b3, (short)0, acc13, false, false);
    }

    // store: rows = mt*16 + 8*khalf + v, col = (cg*4+c)*16 + (lane&15)
    f32x8 r0[4] = {acc00, acc01, acc02, acc03};
    f32x8 r1[4] = {acc10, acc11, acc12, acc13};
    float* c0 = C + (size_t)((mt0 << 4) + (khalf << 3)) * Oc + (cg << 6) + (lane & 15);
    float* c1 = C + (size_t)((mt1 << 4) + (khalf << 3)) * Oc + (cg << 6) + (lane & 15);
#pragma unroll
    for (int c = 0; c < 4; ++c) {
        int col = (cg << 6) + (c << 4) + (lane & 15);
        float bv = bias ? bias[col] : 0.0f;
#pragma unroll
        for (int v = 0; v < 8; ++v) {
            c0[(size_t)v * Oc + (c << 4)] = r0[c][v] + bv;
            if (has1) c1[(size_t)v * Oc + (c << 4)] = r1[c][v] + bv;
        }
    }
}

// ---------------------------------------------------------------------------
// BatchNorm batch statistics per column -> fused scale/shift
// ---------------------------------------------------------------------------
__global__ void bn_stats_kernel(const float* __restrict__ x,
                                const float* __restrict__ g,
                                const float* __restrict__ b,
                                float* __restrict__ scale, float* __restrict__ shift,
                                int N, int F) {
    __shared__ float ssum[256];
    __shared__ float ssq[256];
    int col = blockIdx.x;
    float s = 0.0f, q = 0.0f;
    for (int r = threadIdx.x; r < N; r += blockDim.x) {
        float v = x[(size_t)r * F + col];
        s += v; q += v * v;
    }
    ssum[threadIdx.x] = s; ssq[threadIdx.x] = q;
    __syncthreads();
    for (int st = 128; st > 0; st >>= 1) {
        if ((int)threadIdx.x < st) {
            ssum[threadIdx.x] += ssum[threadIdx.x + st];
            ssq[threadIdx.x]  += ssq[threadIdx.x + st];
        }
        __syncthreads();
    }
    if (threadIdx.x == 0) {
        float inv = 1.0f / (float)N;
        float mu  = ssum[0] * inv;
        float var = ssq[0] * inv - mu * mu;
        float sc  = g[col] * rsqrtf(var + 1e-5f);
        scale[col] = sc;
        shift[col] = b[col] - mu * sc;
    }
}

// ---------------------------------------------------------------------------
// Graph helpers
// ---------------------------------------------------------------------------
__global__ void fill_f32_kernel(float* __restrict__ p, float v, int total) {
    int i = blockIdx.x * blockDim.x + threadIdx.x;
    if (i < total) p[i] = v;
}

__global__ void deg_accum_kernel(const int* __restrict__ dst, float* __restrict__ deg, int E) {
    int e = blockIdx.x * blockDim.x + threadIdx.x;
    if (e < E) atomicAdd(&deg[dst[e]], 1.0f);
}

__global__ void rsqrt_inplace_kernel(float* __restrict__ p, int n) {
    int i = blockIdx.x * blockDim.x + threadIdx.x;
    if (i < n) p[i] = rsqrtf(p[i]);
}

// One block per edge, one thread per feature (F == blockDim.x)
__global__ void edge_agg_kernel(const float* __restrict__ h,
                                const int* __restrict__ src, const int* __restrict__ dst,
                                const float* __restrict__ dis,
                                float* __restrict__ agg, int E, int F) {
    int e = blockIdx.x;
    if (e >= E) return;
    int s = src[e], d = dst[e];
    float w = dis[s] * dis[d];
    int f = threadIdx.x;
    atomicAdd(&agg[(size_t)d * F + f], h[(size_t)s * F + f] * w);
}

// agg = agg + h * dis^2 + bias   (GCN self-loop + bias, in place)
__global__ void gcn_combine_kernel(float* __restrict__ agg, const float* __restrict__ h,
                                   const float* __restrict__ dis,
                                   const float* __restrict__ bias, int total, int F) {
    int i = blockIdx.x * blockDim.x + threadIdx.x;
    if (i >= total) return;
    int node = i / F, f = i - node * F;
    float d = dis[node];
    agg[i] = agg[i] + h[i] * d * d + bias[f];
}

// out = (BN(x_in) + h2c) * 2^-0.5 + x_in    (BN via fused scale/shift)
__global__ void residual_kernel(const float* __restrict__ x_in,
                                const float* __restrict__ scale, const float* __restrict__ shift,
                                const float* __restrict__ h2c,
                                float* __restrict__ out, int total, int F) {
    int i = blockIdx.x * blockDim.x + threadIdx.x;
    if (i >= total) return;
    int f = i % F;
    float xv = x_in[i] * scale[f] + shift[f];
    out[i] = (xv + h2c[i]) * 0.70710678118654752f + x_in[i];
}

// ---------------------------------------------------------------------------
// Host orchestration
// ---------------------------------------------------------------------------
extern "C" void kernel_launch(void* const* d_in, const int* in_sizes, int n_in,
                              void* d_out, int out_size, void* d_ws, size_t ws_size,
                              hipStream_t stream) {
    const float* x   = (const float*)d_in[0];
    const int*   ei  = (const int*)d_in[1];
    const int N = in_sizes[0] / 128;       // 50000 (divisible by 16)
    const int E = in_sizes[1] / 2;         // 800000
    const int* src = ei;
    const int* dst = ei + E;

    const float* g1_lin_W = (const float*)d_in[2];  const float* g1_lin_b = (const float*)d_in[3];
    const float* g1_c1_W  = (const float*)d_in[4];  const float* g1_c1_b  = (const float*)d_in[5];
    const float* g1_c2_W  = (const float*)d_in[6];  const float* g1_c2_b  = (const float*)d_in[7];
    const float* g1_n1_g  = (const float*)d_in[8];  const float* g1_n1_b  = (const float*)d_in[9];
    const float* g1_n2_g  = (const float*)d_in[10]; const float* g1_n2_b  = (const float*)d_in[11];
    const float* g2_lin_W = (const float*)d_in[12]; const float* g2_lin_b = (const float*)d_in[13];
    const float* g2_c1_W  = (const float*)d_in[14]; const float* g2_c1_b  = (const float*)d_in[15];
    const float* g2_c2_W  = (const float*)d_in[16]; const float* g2_c2_b  = (const float*)d_in[17];
    const float* g2_n1_g  = (const float*)d_in[18]; const float* g2_n1_b  = (const float*)d_in[19];
    const float* g2_n2_g  = (const float*)d_in[20]; const float* g2_n2_b  = (const float*)d_in[21];
    const float* out_W    = (const float*)d_in[22]; const float* out_b    = (const float*)d_in[23];
    const float* n1_g     = (const float*)d_in[24]; const float* n1_b     = (const float*)d_in[25];
    const float* l1_W     = (const float*)d_in[26]; const float* l1_b     = (const float*)d_in[27];
    const float* n2_g     = (const float*)d_in[28]; const float* n2_b     = (const float*)d_in[29];
    const float* l2_W     = (const float*)d_in[30]; const float* l2_b     = (const float*)d_in[31];

    // --- workspace carve-out ---
    char* p = (char*)d_ws;
    auto carve = [&](size_t bytes) -> char* {
        char* r = p;
        p += (bytes + 255) & ~(size_t)255;
        return r;
    };
    float*    bufA = (float*)carve((size_t)N * 256 * sizeof(float)); // x_in
    float*    bufB = (float*)carve((size_t)N * 256 * sizeof(float)); // h / x2
    float*    bufC = (float*)carve((size_t)N * 256 * sizeof(float)); // agg / hc
    float*    bufD = (float*)carve((size_t)N * 128 * sizeof(float)); // block2 agg
    _Float16* bufH = (_Float16*)carve((size_t)N * 256 * sizeof(_Float16)); // f16 activations
    float*    dis  = (float*)carve((size_t)N * sizeof(float));
    float* scaleA = (float*)carve(256 * sizeof(float));
    float* shiftA = (float*)carve(256 * sizeof(float));
    float* scaleB = (float*)carve(256 * sizeof(float));
    float* shiftB = (float*)carve(256 * sizeof(float));
    _Float16* wpLin1 = (_Float16*)carve((size_t)128 * 256 * 2);
    _Float16* wpC11  = (_Float16*)carve((size_t)256 * 256 * 2);
    _Float16* wpC21  = (_Float16*)carve((size_t)256 * 256 * 2);
    _Float16* wpLin2 = (_Float16*)carve((size_t)256 * 128 * 2);
    _Float16* wpC12  = (_Float16*)carve((size_t)128 * 128 * 2);
    _Float16* wpC22  = (_Float16*)carve((size_t)128 * 128 * 2);
    _Float16* wpOut  = (_Float16*)carve((size_t)128 * 128 * 2);
    _Float16* wpL1   = (_Float16*)carve((size_t)128 * 64 * 2);
    _Float16* wpL2   = (_Float16*)carve((size_t)64 * 64 * 2);

    auto pack = [&](const float* W, _Float16* Wp, int K, int Oc) {
        int total = K * Oc;
        pack_w_kernel<<<CEIL_DIV(total, 256), 256, 0, stream>>>(W, Wp, K, Oc);
    };
    auto xform = [&](const float* xin, const float* sc, const float* sh, int lrelu, int F) {
        int total = N * F;
        xform_f16_kernel<<<CEIL_DIV(total, 256), 256, 0, stream>>>(xin, sc, sh, lrelu, bufH, total, F);
    };
    auto gemm = [&](const _Float16* Wp, const float* bias, float* C, int K, int Oc) {
        int MT = N >> 4;
        int grps = ((MT + 1) >> 1) * (Oc >> 6);
        gemm_wmma_f16<<<CEIL_DIV(grps, 8), 256, 0, stream>>>(bufH, Wp, bias, C, N, K, Oc);
    };
    auto bn = [&](const float* xin, const float* g, const float* b, float* sc, float* sh, int F) {
        bn_stats_kernel<<<F, 256, 0, stream>>>(xin, g, b, sc, sh, N, F);
    };
    auto zero = [&](float* buf, int total) {
        fill_f32_kernel<<<CEIL_DIV(total, 256), 256, 0, stream>>>(buf, 0.0f, total);
    };
    auto gcn_agg = [&](const float* h, float* agg, const float* bias, int F) {
        zero(agg, N * F);
        edge_agg_kernel<<<E, F, 0, stream>>>(h, src, dst, dis, agg, E, F);
        gcn_combine_kernel<<<CEIL_DIV(N * F, 256), 256, 0, stream>>>(agg, h, dis, bias, N * F, F);
    };

    // --- weight packing (cheap, every call: deterministic) ---
    pack(g1_lin_W, wpLin1, 128, 256);
    pack(g1_c1_W,  wpC11,  256, 256);
    pack(g1_c2_W,  wpC21,  256, 256);
    pack(g2_lin_W, wpLin2, 256, 128);
    pack(g2_c1_W,  wpC12,  128, 128);
    pack(g2_c2_W,  wpC22,  128, 128);
    pack(out_W,    wpOut,  128, 128);
    pack(l1_W,     wpL1,   128, 64);
    pack(l2_W,     wpL2,   64,  64);

    // --- degrees: deg = 1 + count(dst); dis = rsqrt(deg) ---
    fill_f32_kernel<<<CEIL_DIV(N, 256), 256, 0, stream>>>(dis, 1.0f, N);
    deg_accum_kernel<<<CEIL_DIV(E, 256), 256, 0, stream>>>(dst, dis, E);
    rsqrt_inplace_kernel<<<CEIL_DIV(N, 256), 256, 0, stream>>>(dis, N);

    // ===================== Block 1 (128 -> 256) =====================
    xform(x, nullptr, nullptr, 0, 128);
    gemm(wpLin1, g1_lin_b, bufA, 128, 256);                 // x_in = x@W + b
    bn(bufA, g1_n1_g, g1_n1_b, scaleA, shiftA, 256);        // xv = BN(x_in)
    xform(bufA, scaleA, shiftA, 1, 256);                    // lrelu(xv) -> f16
    gemm(wpC11, nullptr, bufB, 256, 256);                   // h = lrelu(xv)@c1W
    gcn_agg(bufB, bufC, g1_c1_b, 256);                      // bufC = GCN out
    bn(bufC, g1_n2_g, g1_n2_b, scaleB, shiftB, 256);
    xform(bufC, scaleB, shiftB, 1, 256);
    gemm(wpC21, nullptr, bufB, 256, 256);
    gcn_agg(bufB, bufC, g1_c2_b, 256);                      // bufC = h (2nd GCN)
    residual_kernel<<<CEIL_DIV(N * 256, 256), 256, 0, stream>>>(
        bufA, scaleA, shiftA, bufC, bufB, N * 256, 256);    // x2 = bufB

    // ===================== Block 2 (256 -> 128) =====================
    xform(bufB, nullptr, nullptr, 0, 256);
    gemm(wpLin2, g2_lin_b, bufA, 256, 128);                 // x_in2
    bn(bufA, g2_n1_g, g2_n1_b, scaleA, shiftA, 128);
    xform(bufA, scaleA, shiftA, 1, 128);
    gemm(wpC12, nullptr, bufC, 128, 128);
    gcn_agg(bufC, bufD, g2_c1_b, 128);
    bn(bufD, g2_n2_g, g2_n2_b, scaleB, shiftB, 128);
    xform(bufD, scaleB, shiftB, 1, 128);
    gemm(wpC22, nullptr, bufC, 128, 128);
    gcn_agg(bufC, bufD, g2_c2_b, 128);
    residual_kernel<<<CEIL_DIV(N * 128, 256), 256, 0, stream>>>(
        bufA, scaleA, shiftA, bufD, bufB, N * 128, 128);    // x3 = bufB

    // ===================== Head =====================
    xform(bufB, nullptr, nullptr, 0, 128);
    gemm(wpOut, out_b, bufA, 128, 128);                     // h = x3@out_W + out_b
    bn(bufA, n1_g, n1_b, scaleA, shiftA, 128);
    xform(bufA, scaleA, shiftA, 1, 128);
    gemm(wpL1, l1_b, bufC, 128, 64);                        // h = lrelu(BN(h))@l1 + b
    bn(bufC, n2_g, n2_b, scaleB, shiftB, 64);
    xform(bufC, scaleB, shiftB, 1, 64);
    gemm(wpL2, l2_b, (float*)d_out, 64, 64);                // out = lrelu(BN(h))@l2 + b
}
